// SS2D_43353399886013
// MI455X (gfx1250) — matrix operand order repaired and checked by reading, not compiled, not run on hardware
//
#include <hip/hip_runtime.h>
#include <hip/hip_bf16.h>

typedef __attribute__((ext_vector_type(16))) __bf16 v16bf;
typedef __attribute__((ext_vector_type(8)))  float  v8f;
typedef __attribute__((ext_vector_type(4)))  unsigned int u32x4;
typedef __attribute__((ext_vector_type(8)))  int i32x8;
typedef __attribute__((ext_vector_type(4)))  int i32x4;

#define B_   4
#define H_   64
#define W_   64
#define C_   96
#define DI_  192
#define NST  16
#define KDIR 4
#define RANK 6
#define L_   4096
#define CPROJ 38   /* RANK + 2*NST */
#define CPAD  48   /* padded to 3 WMMA M-tiles */
#define LCH  64    /* scan chunk length */
#define CH   64    /* number of chunks = L_/LCH */

#if defined(__has_builtin)
#if __has_builtin(__builtin_amdgcn_tensor_load_to_lds) && \
    __has_builtin(__builtin_amdgcn_s_wait_tensorcnt)
#define HAVE_TDM 1
#endif
#endif
#ifndef HAVE_TDM
#define HAVE_TDM 0
#endif

// Native f32->bf16 convert: plain fptrunc lets the backend select the
// hardware packed convert on gfx1250 instead of manual bit-twiddling.
static __device__ __forceinline__ __bf16 f2bf(float f) {
    return (__bf16)f;
}

// scan-order index map: value of xs[k][d][l] lives at xconv[d][perm_idx(k,l)]
static __device__ __forceinline__ int perm_idx(int k, int l) {
    if (k & 2) l = (L_ - 1) - l;                 // flipped directions
    if (k & 1) l = ((l & 63) << 6) | (l >> 6);   // transpose: l = w*H + h -> h*W + w
    return l;
}

// ---------------------------------------------------------------------------
// TDM 2D tile load (rows x cols f32, row stride in elements) into LDS.
// D# layout per CDNA5 ISA. Issued by one wave;
// EXEC is ignored by TENSOR_LOAD_TO_LDS so gating must be wave-uniform.
// This toolchain exposes the 6-arg builtin:
//   (uint32x4 g0, int32x8 g1, int32x4 g2, int32x4 g3, int32x8 gx, i32 cpol)
// ---------------------------------------------------------------------------
#if HAVE_TDM
static __device__ __forceinline__ void tdm_load_tile_f32(
    unsigned lds_off, const float* gptr, int rows, int cols, unsigned stride_elems) {
    unsigned long long ga = (unsigned long long)(__UINTPTR_TYPE__)gptr;
    u32x4 g0;
    g0.x = 1u;                                            // count=1, user mode
    g0.y = lds_off;                                       // LDS byte address
    g0.z = (unsigned)(ga & 0xFFFFFFFFu);                  // global_addr[31:0]
    g0.w = (unsigned)((ga >> 32) & 0x01FFFFFFu) | (2u << 30); // [56:32] | type=2
    const unsigned td0 = 1u << 20, td1 = 1u << 20;        // generous OOB dims
    i32x8 g1;
    g1[0] = (int)(2u << 16);                              // data_size = 4 bytes
    g1[1] = (int)((td0 & 0xFFFFu) << 16);                 // tensor_dim0[15:0]
    g1[2] = (int)((td0 >> 16) | ((td1 & 0xFFFFu) << 16)); // td0 hi | td1 lo
    g1[3] = (int)((td1 >> 16) | ((unsigned)cols << 16));  // td1 hi | tile_dim0
    g1[4] = (int)(unsigned)rows;                          // tile_dim1, tile_dim2=0
    g1[5] = (int)stride_elems;                            // tensor_dim0_stride lo
    g1[6] = 0;                                            // stride hi | dim1_stride lo
    g1[7] = 0;
    i32x4 z4 = {0, 0, 0, 0};
    i32x8 z8 = {0, 0, 0, 0, 0, 0, 0, 0};
    __builtin_amdgcn_tensor_load_to_lds(g0, g1, z4, z4, z8, 0);
}
#endif

// Stage a NST x LCH f32 tile (row stride L_) into smem[NST][LCH].
// Caller must __syncthreads() afterwards. blockDim.x == 192.
static __device__ __forceinline__ void stage_tile(float* smem,
                                                  const float* gsrc, int l0) {
#if HAVE_TDM
    if (threadIdx.x < 32) {                       // wave-uniform: wave 0 only
        tdm_load_tile_f32((unsigned)(__UINTPTR_TYPE__)smem, gsrc + l0, NST, LCH, L_);
        __builtin_amdgcn_s_wait_tensorcnt(0);
    }
#else
    for (int idx = threadIdx.x; idx < NST * LCH; idx += 192) {
        int n = idx / LCH, lc = idx % LCH;
        smem[n * LCH + lc] = gsrc[(size_t)n * L_ + l0 + lc];
    }
#endif
}

// ---------------------------------------------------------------------------
// Kernel 1: xz = X(16384x96) @ in_proj_w^T(96x384), bf16 WMMA, f32 accum
// ---------------------------------------------------------------------------
__global__ __launch_bounds__(256) void k_in_proj(const float* __restrict__ X,
                                                 const float* __restrict__ W1,
                                                 float* __restrict__ XZ) {
    const int lane = threadIdx.x & 31;
    const int wave = threadIdx.x >> 5;
    const int tile = blockIdx.x * 8 + wave;      // 1024 * 24 wave-tiles
    const int mt = tile / 24, nt = tile % 24;
    const int half = lane >> 4;
    const int mrow = mt * 16 + (lane & 15);
    const int ncol = nt * 16 + (lane & 15);
    v8f acc = {};
    for (int k0 = 0; k0 < C_; k0 += 32) {
        v16bf a, b;
#pragma unroll
        for (int i = 0; i < 16; ++i) {
            int ka = k0 + ((i < 8) ? i : (i + 8)) + half * 8;   // A 16x32 layout
            a[i] = f2bf(X[mrow * C_ + ka]);
            int kb = k0 + ((i < 8) ? i : (i + 8)) + half * 8;   // B 32x16 layout
            b[i] = f2bf(W1[ncol * C_ + kb]);                    // W1 is (384,96) row-major
        }
        acc = __builtin_amdgcn_wmma_f32_16x16x32_bf16(false, a, false, b,
                                                      (short)0, acc, false, false);
    }
#pragma unroll
    for (int r = 0; r < 8; ++r) {
        int m = mt * 16 + half * 8 + r;
        XZ[m * 384 + nt * 16 + (lane & 15)] = acc[r];
    }
}

// ---------------------------------------------------------------------------
// Kernel 2: depthwise 3x3 conv (SAME) + bias + SiLU; xz(B,L,384)->xconv(B,Di,L)
// ---------------------------------------------------------------------------
__global__ __launch_bounds__(256) void k_conv(const float* __restrict__ XZ,
                                              const float* __restrict__ CW,
                                              const float* __restrict__ CB,
                                              float* __restrict__ XC) {
    int tid = blockIdx.x * 256 + threadIdx.x;    // B*Di*L threads
    int l = tid & (L_ - 1);
    int d = (tid >> 12) % DI_;
    int bb = tid / (L_ * DI_);
    int h = l >> 6, w = l & 63;
    float sum = CB[d];
#pragma unroll
    for (int dy = 0; dy < 3; ++dy) {
#pragma unroll
        for (int dx = 0; dx < 3; ++dx) {
            int hh = h + dy - 1, ww = w + dx - 1;
            if (hh >= 0 && hh < H_ && ww >= 0 && ww < W_) {
                sum += CW[d * 9 + dy * 3 + dx] *
                       XZ[((size_t)(bb * L_ + hh * W_ + ww)) * 384 + d];
            }
        }
    }
    float s = sum / (1.f + __expf(-sum));        // SiLU
    XC[((size_t)(bb * DI_ + d)) * L_ + l] = s;
}

// ---------------------------------------------------------------------------
// Kernel 3: x_dbl = x_proj_w[k](38x192) @ xs[b,k](192x4096), WMMA, M padded to 48
// ---------------------------------------------------------------------------
__global__ __launch_bounds__(256) void k_xdbl(const float* __restrict__ XC,
                                              const float* __restrict__ XPW,
                                              float* __restrict__ XD) {
    const int lane = threadIdx.x & 31;
    const int wave = threadIdx.x >> 5;
    int t = blockIdx.x * 8 + wave;               // B*K*3*256 wave-tiles
    const int nt = t & 255;  t >>= 8;
    const int mt = t % 3;    t /= 3;
    const int k  = t & 3;
    const int bb = t >> 2;
    const int half = lane >> 4;
    const int m = mt * 16 + (lane & 15);
    const int n = nt * 16 + (lane & 15);
    const int src = perm_idx(k, n);              // scan-order gather, shared over K
    v8f acc = {};
    for (int k0 = 0; k0 < DI_; k0 += 32) {
        v16bf a, b;
#pragma unroll
        for (int i = 0; i < 16; ++i) {
            int ka = k0 + ((i < 8) ? i : (i + 8)) + half * 8;
            a[i] = (m < CPROJ) ? f2bf(XPW[(k * CPROJ + m) * DI_ + ka]) : f2bf(0.f);
            int dd = k0 + ((i < 8) ? i : (i + 8)) + half * 8;
            b[i] = f2bf(XC[((size_t)(bb * DI_ + dd)) * L_ + src]);
        }
        acc = __builtin_amdgcn_wmma_f32_16x16x32_bf16(false, a, false, b,
                                                      (short)0, acc, false, false);
    }
#pragma unroll
    for (int r = 0; r < 8; ++r) {
        int row = mt * 16 + half * 8 + r;
        XD[((size_t)((bb * KDIR + k) * CPAD + row)) * L_ + nt * 16 + (lane & 15)] = acc[r];
    }
}

// ---------------------------------------------------------------------------
// Kernel 4: delta = softplus(dts @ dt_projs_w^T + dt_projs_b)   (RANK=6: VALU)
// ---------------------------------------------------------------------------
__global__ __launch_bounds__(256) void k_delta(const float* __restrict__ XD,
                                               const float* __restrict__ DTW,
                                               const float* __restrict__ DTB,
                                               float* __restrict__ DEL) {
    int tid = blockIdx.x * 256 + threadIdx.x;    // B*K*Di*L threads
    int l = tid & (L_ - 1);
    int rest = tid >> 12;
    int d = rest % DI_;
    int bk = rest / DI_;
    int k = bk & 3;
    float acc = DTB[k * DI_ + d];
#pragma unroll
    for (int r = 0; r < RANK; ++r)
        acc += XD[((size_t)(bk * CPAD + r)) * L_ + l] * DTW[(k * DI_ + d) * RANK + r];
    float sp = (acc > 20.f) ? acc : log1pf(__expf(acc));
    DEL[((size_t)(bk * DI_ + d)) * L_ + l] = sp;
}

// ---------------------------------------------------------------------------
// Kernel 5a: chunk-local scans. One block per (b,k,chunk): 192 lanes (one per d).
// h <- exp(dt*A)*h + dt*u*B with h0=0; also accumulates sum(dt) so the chunk
// transition matrix collapses to exp(A * sum_dt). B tile staged via TDM.
// ---------------------------------------------------------------------------
__global__ __launch_bounds__(192) void k_scan_part(const float* __restrict__ XC,
                                                   const float* __restrict__ XD,
                                                   const float* __restrict__ DEL,
                                                   const float* __restrict__ ALOG,
                                                   float* __restrict__ PH,
                                                   float* __restrict__ SD) {
    const int blk = blockIdx.x;                  // bk*CH + c
    const int c  = blk & (CH - 1);
    const int bk = blk >> 6;
    const int bb = bk >> 2, k = bk & 3;
    const int d = threadIdx.x;
    const int l0 = c * LCH;

    __shared__ float sB[NST][LCH];
    stage_tile(&sB[0][0], XD + ((size_t)(bk * CPAD + RANK)) * L_, l0);

    float A[NST], hs[NST];
#pragma unroll
    for (int n = 0; n < NST; ++n) {
        A[n] = -__expf(ALOG[(k * DI_ + d) * NST + n]);
        hs[n] = 0.f;
    }
    const float* del = DEL + ((size_t)(bk * DI_ + d)) * L_;
    const float* xc  = XC + ((size_t)(bb * DI_ + d)) * L_;
    float sdt = 0.f;
    __syncthreads();
#pragma unroll 4
    for (int j = 0; j < LCH; ++j) {
        int l = l0 + j;
        float dt = del[l];
        float u  = xc[perm_idx(k, l)];
        float du = dt * u;
        sdt += dt;
#pragma unroll
        for (int n = 0; n < NST; ++n)
            hs[n] = hs[n] * __expf(dt * A[n]) + du * sB[n][j];
    }
    size_t po = ((size_t)(bk * DI_ + d) * CH + c) * NST;
#pragma unroll
    for (int n = 0; n < NST; ++n) PH[po + n] = hs[n];
    SD[(size_t)(bk * DI_ + d) * CH + c] = sdt;
}

// ---------------------------------------------------------------------------
// Kernel 5b: serial combine over 64 chunk summaries per chain (3072 chains).
// Rewrites PH[c] with the prefix state entering chunk c.
// ---------------------------------------------------------------------------
__global__ __launch_bounds__(256) void k_scan_combine(const float* __restrict__ ALOG,
                                                      float* __restrict__ PH,
                                                      const float* __restrict__ SD) {
    int tid = blockIdx.x * 256 + threadIdx.x;    // 3072 threads exactly
    int d = tid % DI_;
    int bk = tid / DI_;
    int k = bk & 3;
    float A[NST], h[NST];
#pragma unroll
    for (int n = 0; n < NST; ++n) {
        A[n] = -__expf(ALOG[(k * DI_ + d) * NST + n]);
        h[n] = 0.f;
    }
    for (int c = 0; c < CH; ++c) {
        float sdt = SD[(size_t)(bk * DI_ + d) * CH + c];
        size_t po = ((size_t)(bk * DI_ + d) * CH + c) * NST;
#pragma unroll
        for (int n = 0; n < NST; ++n) {
            float hl = PH[po + n];               // chunk-local terminal state
            float hp = h[n];                     // prefix entering chunk c
            PH[po + n] = hp;
            h[n] = hp * __expf(A[n] * sdt) + hl;
        }
    }
}

// ---------------------------------------------------------------------------
// Kernel 5c: replay each chunk from its true prefix, emit y (+ D*u).
// B and C tiles staged via TDM.
// ---------------------------------------------------------------------------
__global__ __launch_bounds__(192) void k_scan_final(const float* __restrict__ XC,
                                                    const float* __restrict__ XD,
                                                    const float* __restrict__ DEL,
                                                    const float* __restrict__ ALOG,
                                                    const float* __restrict__ DS,
                                                    const float* __restrict__ PH,
                                                    float* __restrict__ YS) {
    const int blk = blockIdx.x;
    const int c  = blk & (CH - 1);
    const int bk = blk >> 6;
    const int bb = bk >> 2, k = bk & 3;
    const int d = threadIdx.x;
    const int l0 = c * LCH;

    __shared__ float sB[NST][LCH];
    __shared__ float sC[NST][LCH];
#if HAVE_TDM
    if (threadIdx.x < 32) {                      // wave 0: two TDM ops, one wait
        tdm_load_tile_f32((unsigned)(__UINTPTR_TYPE__)&sB[0][0],
                          XD + ((size_t)(bk * CPAD + RANK)) * L_ + l0, NST, LCH, L_);
        tdm_load_tile_f32((unsigned)(__UINTPTR_TYPE__)&sC[0][0],
                          XD + ((size_t)(bk * CPAD + RANK + NST)) * L_ + l0, NST, LCH, L_);
        __builtin_amdgcn_s_wait_tensorcnt(0);
    }
#else
    stage_tile(&sB[0][0], XD + ((size_t)(bk * CPAD + RANK)) * L_, l0);
    stage_tile(&sC[0][0], XD + ((size_t)(bk * CPAD + RANK + NST)) * L_, l0);
#endif

    float A[NST], hs[NST];
    size_t po = ((size_t)(bk * DI_ + d) * CH + c) * NST;
#pragma unroll
    for (int n = 0; n < NST; ++n) {
        A[n] = -__expf(ALOG[(k * DI_ + d) * NST + n]);
        hs[n] = PH[po + n];                      // prefix state entering chunk
    }
    const float Dd = DS[k * DI_ + d];
    const float* del = DEL + ((size_t)(bk * DI_ + d)) * L_;
    const float* xc  = XC + ((size_t)(bb * DI_ + d)) * L_;
    float* ys        = YS + ((size_t)(bk * DI_ + d)) * L_;
    __syncthreads();
#pragma unroll 4
    for (int j = 0; j < LCH; ++j) {
        int l = l0 + j;
        float dt = del[l];
        float u  = xc[perm_idx(k, l)];
        float du = dt * u;
        float y = 0.f;
#pragma unroll
        for (int n = 0; n < NST; ++n) {
            hs[n] = hs[n] * __expf(dt * A[n]) + du * sB[n][j];
            y += hs[n] * sC[n][j];
        }
        ys[l] = y + Dd * u;
    }
}

// ---------------------------------------------------------------------------
// Kernel 6: merge 4 directions + LayerNorm(Di) + SiLU(z) gate -> yg(B,L,Di)
// ---------------------------------------------------------------------------
__global__ __launch_bounds__(192) void k_merge(const float* __restrict__ YS,
                                               const float* __restrict__ XZ,
                                               const float* __restrict__ LG,
                                               const float* __restrict__ LB,
                                               float* __restrict__ YG) {
    const int bl = blockIdx.x;
    const int bb = bl >> 12, l = bl & (L_ - 1);
    const int d = threadIdx.x;
    const int h = l >> 6, w = l & 63;
    const int lt = (w << 6) | h;                 // index in transposed scan order
    const size_t base = (size_t)bb * KDIR * DI_ * L_;
    const size_t row = (size_t)d * L_;
    float y = YS[base + 0 * (size_t)DI_ * L_ + row + l]
            + YS[base + 2 * (size_t)DI_ * L_ + row + (L_ - 1 - l)]
            + YS[base + 1 * (size_t)DI_ * L_ + row + lt]
            + YS[base + 3 * (size_t)DI_ * L_ + row + (L_ - 1 - lt)];

    __shared__ float s1[256], s2[256];
    s1[d] = y; s2[d] = y * y;
    if (d < 64) { s1[DI_ + d] = 0.f; s2[DI_ + d] = 0.f; }
    __syncthreads();
    for (int s = 128; s > 0; s >>= 1) {
        if (d < s) { s1[d] += s1[d + s]; s2[d] += s2[d + s]; }
        __syncthreads();
    }
    float mean = s1[0] * (1.f / DI_);
    float var  = s2[0] * (1.f / DI_) - mean * mean;
    float yn = (y - mean) * rsqrtf(var + 1e-5f) * LG[d] + LB[d];
    float z = XZ[((size_t)(bb * L_ + l)) * 384 + DI_ + d];
    float sz = z / (1.f + __expf(-z));
    YG[((size_t)(bb * L_ + l)) * DI_ + d] = yn * sz;
}

// ---------------------------------------------------------------------------
// Kernel 7: out = YG(16384x192) @ out_proj_w^T(192x96), WMMA
// ---------------------------------------------------------------------------
__global__ __launch_bounds__(256) void k_out_proj(const float* __restrict__ YG,
                                                  const float* __restrict__ WO,
                                                  float* __restrict__ OUT) {
    const int lane = threadIdx.x & 31;
    const int wave = threadIdx.x >> 5;
    const int tile = blockIdx.x * 8 + wave;      // 1024 * 6 wave-tiles
    const int mt = tile / 6, nt = tile % 6;
    const int half = lane >> 4;
    const int mrow = mt * 16 + (lane & 15);
    const int ncol = nt * 16 + (lane & 15);
    v8f acc = {};
    for (int k0 = 0; k0 < DI_; k0 += 32) {
        v16bf a, b;
#pragma unroll
        for (int i = 0; i < 16; ++i) {
            int ka = k0 + ((i < 8) ? i : (i + 8)) + half * 8;
            a[i] = f2bf(YG[(size_t)mrow * DI_ + ka]);
            int kb = k0 + ((i < 8) ? i : (i + 8)) + half * 8;
            b[i] = f2bf(WO[ncol * DI_ + kb]);    // WO is (96,192) row-major
        }
        acc = __builtin_amdgcn_wmma_f32_16x16x32_bf16(false, a, false, b,
                                                      (short)0, acc, false, false);
    }
#pragma unroll
    for (int r = 0; r < 8; ++r) {
        int m = mt * 16 + half * 8 + r;
        OUT[(size_t)m * C_ + nt * 16 + (lane & 15)] = acc[r];
    }
}

// ---------------------------------------------------------------------------
extern "C" void kernel_launch(void* const* d_in, const int* in_sizes, int n_in,
                              void* d_out, int out_size, void* d_ws, size_t ws_size,
                              hipStream_t stream) {
    (void)in_sizes; (void)n_in; (void)out_size; (void)ws_size;
    const float* x          = (const float*)d_in[0];
    const float* in_proj_w  = (const float*)d_in[1];
    const float* conv_w     = (const float*)d_in[2];
    const float* conv_b     = (const float*)d_in[3];
    const float* x_proj_w   = (const float*)d_in[4];
    const float* dt_projs_w = (const float*)d_in[5];
    const float* dt_projs_b = (const float*)d_in[6];
    const float* A_logs     = (const float*)d_in[7];
    const float* Ds         = (const float*)d_in[8];
    const float* ln_g       = (const float*)d_in[9];
    const float* ln_b       = (const float*)d_in[10];
    const float* out_proj_w = (const float*)d_in[11];
    float* out = (float*)d_out;

    float* ws    = (float*)d_ws;
    float* xz    = ws;                                    // B*L*384
    float* xconv = xz    + (size_t)B_ * L_ * 384;         // B*Di*L
    float* xdbl  = xconv + (size_t)B_ * DI_ * L_;         // B*K*48*L
    float* delta = xdbl  + (size_t)B_ * KDIR * CPAD * L_; // B*K*Di*L
    float* ys    = delta + (size_t)B_ * KDIR * DI_ * L_;  // B*K*Di*L
    float* yg    = ys    + (size_t)B_ * KDIR * DI_ * L_;  // B*L*Di
    float* ph    = yg    + (size_t)B_ * L_ * DI_;         // B*K*Di*CH*NST
    float* sd    = ph    + (size_t)B_ * KDIR * DI_ * CH * NST; // B*K*Di*CH

    // 1. in_proj GEMM: 1024 M-tiles x 24 N-tiles, 8 wave-tiles/block
    k_in_proj<<<3072, 256, 0, stream>>>(x, in_proj_w, xz);
    // 2. depthwise conv + SiLU
    k_conv<<<(B_ * DI_ * L_) / 256, 256, 0, stream>>>(xz, conv_w, conv_b, xconv);
    // 3. x_dbl GEMMs: B*K*3*256 wave-tiles
    k_xdbl<<<(B_ * KDIR * 3 * 256) / 8, 256, 0, stream>>>(xconv, x_proj_w, xdbl);
    // 4. dt projection + softplus
    k_delta<<<(B_ * KDIR * DI_ * L_) / 256, 256, 0, stream>>>(xdbl, dt_projs_w,
                                                              dt_projs_b, delta);
    // 5. chunk-parallel selective scan
    k_scan_part<<<B_ * KDIR * CH, DI_, 0, stream>>>(xconv, xdbl, delta, A_logs, ph, sd);
    k_scan_combine<<<(B_ * KDIR * DI_) / 256, 256, 0, stream>>>(A_logs, ph, sd);
    k_scan_final<<<B_ * KDIR * CH, DI_, 0, stream>>>(xconv, xdbl, delta, A_logs,
                                                     Ds, ph, ys);
    // 6. merge + LayerNorm + gate
    k_merge<<<B_ * L_, DI_, 0, stream>>>(ys, xz, ln_g, ln_b, yg);
    // 7. out_proj GEMM: 1024 x 6 wave-tiles
    k_out_proj<<<(1024 * 6) / 8, 256, 0, stream>>>(yg, out_proj_w, out);
}
